// AttentionDot_3882650435656
// MI455X (gfx1250) — matrix operand order repaired and checked by reading, not compiled, not run on hardware
//
#include <hip/hip_runtime.h>
#include <stdint.h>

// ---------------------------------------------------------------------------
// Fused batch-softmax attention for MI455X (gfx1250, wave32, WMMA).
//   scores = K @ Q^T              (per batch)
//   w      = softmax(softmax(scores, axis=batch), axis=batch)   <- size-4 axis!
//   out    = w @ V
// The softmax axis is the batch axis (elementwise in n,m) so the whole pipe
// fuses; never materialize the [4,4096,4096] score tensor (256 MB).
// ~34.4 GFLOP vs ~34 MB I/O -> compute bound -> bf16 WMMA (16x16x32), f32 acc.
// ---------------------------------------------------------------------------

typedef __attribute__((ext_vector_type(16))) __bf16 v16bf;
typedef __attribute__((ext_vector_type(8)))  float  v8f;
typedef uint32_t u32;
typedef __attribute__((ext_vector_type(4))) u32 u32x4;
typedef __attribute__((ext_vector_type(8))) u32 u32x8;

#define BATCH   4
#define SEQ_N   4096
#define SEQ_M   4096
#define DIM     128
#define NT      16      // n rows per block
#define MT      32      // m columns per iteration (one WMMA K-step)
#define THREADS 128     // 4 wave32s

union FragBF { v16bf v; uint4 q2[2]; __bf16 e[16]; };
union FragF  { v8f  v; float e[8]; };

__device__ __forceinline__ u32 lds_addr32(const void* p) {
  // generic shared pointer: low 32 bits are the LDS byte offset (aperture map)
  return (u32)(uintptr_t)p;
}

// A-matrix fragment (16-bit, 16xK=32) from row-major bf16 LDS tile.
// Layout: lane L holds row M=L%16; element s: K = kb + s,
// kb = (L<16 ? 0 : 16). K-contiguous => 2x b128.
__device__ __forceinline__ v16bf afrag_lds(const __bf16* base, int ld,
                                           int row, int k0, int kb) {
  FragBF f;
  f.q2[0] = *(const uint4*)(base + row * ld + k0 + kb);
  f.q2[1] = *(const uint4*)(base + row * ld + k0 + kb + 8);
  return f.v;
}

__global__ __launch_bounds__(THREADS)
void attn_batchsoftmax_kernel(const float* __restrict__ kin,
                              const float* __restrict__ qin,
                              const float* __restrict__ vin,
                              float* __restrict__ out) {
  __shared__ __align__(16) __bf16 Kt[BATCH * NT * DIM];  // 16 KB  K tiles bf16
  __shared__ __align__(16) float  Sb[BATCH * NT * MT];   //  8 KB  raw scores
  __shared__ __align__(16) __bf16 Wt[BATCH * NT * MT];   //  4 KB  softmax wts
  __shared__ __align__(16) float  Vst[MT * DIM];         // 16 KB  V tile stage

  const int tid  = threadIdx.x;
  const int w    = tid >> 5;        // wave id 0..3
  const int lane = tid & 31;
  const int r    = lane & 15;       // row (A) / col (B,C) within 16
  const int hb   = lane >> 4;       // lane half
  const int kbA  = hb * 16;         // A-frag K base per lane half
  const int kbB  = hb * 16;         // B-frag K base per lane half
  const int n0   = blockIdx.x * NT;

  // ---- prologue: TDM-load K tiles (fp32) into LDS, convert to bf16 ----
  for (int b = 0; b < BATCH; ++b) {
    if (w == 0) {
      const uint64_t ga = (uint64_t)(uintptr_t)(kin + ((size_t)b * SEQ_N + n0) * DIM);
      u32x4 g0;
      g0[0] = 1u;                                   // count=1, user descriptor
      g0[1] = lds_addr32(Vst);                      // lds_addr
      g0[2] = (u32)ga;                              // global_addr lo
      g0[3] = (u32)((ga >> 32) & 0x1FFFFFFu) | (2u << 30); // addr hi | type=2
      u32x8 g1;
      g1[0] = 2u << 16;                             // data_size = 4B
      g1[1] = ((u32)DIM & 0xFFFFu) << 16;           // tensor_dim0 lo16
      g1[2] = ((u32)DIM >> 16) | (((u32)SEQ_N & 0xFFFFu) << 16); // dim0 hi | dim1 lo
      g1[3] = ((u32)SEQ_N >> 16) | ((u32)DIM << 16);             // dim1 hi | tile_dim0
      g1[4] = (u32)NT;                              // tile_dim1 (tile_dim2=0)
      g1[5] = (u32)DIM;                             // tensor_dim0_stride lo32
      g1[6] = 0u;
      g1[7] = 0u;
      asm volatile("tensor_load_to_lds %0, %1" :: "s"(g0), "s"(g1) : "memory");
      __builtin_amdgcn_s_wait_tensorcnt(0);
    }
    __syncthreads();
    #pragma unroll
    for (int j = 0; j < 16; ++j) {
      int idx = tid * 16 + j;                       // 2048 elems / 128 thr
      Kt[b * (NT * DIM) + idx] = (__bf16)Vst[idx];
    }
    __syncthreads();
  }

  // ---- output accumulators: [batch][2 x 16-d-col chunks], f32 C layout ----
  FragF oacc[BATCH][2];
  #pragma unroll
  for (int b = 0; b < BATCH; ++b)
    #pragma unroll
    for (int c = 0; c < 2; ++c)
      #pragma unroll
      for (int ee = 0; ee < 8; ++ee) oacc[b][c].e[ee] = 0.0f;

  for (int m0 = 0; m0 < SEQ_M; m0 += MT) {
    // ================= GEMM1: wave w computes scores for batch w ==========
    FragF sacc[2];
    #pragma unroll
    for (int c = 0; c < 2; ++c)
      #pragma unroll
      for (int ee = 0; ee < 8; ++ee) sacc[c].e[ee] = 0.0f;

    #pragma unroll
    for (int dc = 0; dc < 4; ++dc) {                // K-dim in chunks of 32
      v16bf a = afrag_lds(Kt + w * (NT * DIM), DIM, r, dc * 32, kbA);
      #pragma unroll
      for (int mc = 0; mc < 2; ++mc) {              // two 16-col score tiles
        // B frag: lane holds col N=r, K=d -> 16 contiguous floats of q row
        const float* qp = qin + ((size_t)w * SEQ_M + m0 + mc * 16 + r) * DIM
                              + dc * 32 + kbB;
        float4 f0 = ((const float4*)qp)[0];
        float4 f1 = ((const float4*)qp)[1];
        float4 f2 = ((const float4*)qp)[2];
        float4 f3 = ((const float4*)qp)[3];
        float t[16] = {f0.x, f0.y, f0.z, f0.w, f1.x, f1.y, f1.z, f1.w,
                       f2.x, f2.y, f2.z, f2.w, f3.x, f3.y, f3.z, f3.w};
        FragBF fb;
        #pragma unroll
        for (int ee = 0; ee < 16; ++ee) fb.e[ee] = (__bf16)t[ee];
        sacc[mc].v = __builtin_amdgcn_wmma_f32_16x16x32_bf16(
            false, a, false, fb.v, (short)0, sacc[mc].v, false, false);
      }
    }
    // C layout: VGPR rr -> row rr+8*hb, col r
    #pragma unroll
    for (int mc = 0; mc < 2; ++mc)
      #pragma unroll
      for (int rr = 0; rr < 8; ++rr)
        Sb[w * (NT * MT) + (rr + 8 * hb) * MT + mc * 16 + r] = sacc[mc].e[rr];
    __syncthreads();

    // ============ double softmax over the size-4 batch axis ===============
    #pragma unroll
    for (int i = 0; i < 4; ++i) {                   // 512 positions / 128 thr
      int p = tid + i * THREADS;
      float s0 = Sb[p], s1 = Sb[512 + p], s2 = Sb[1024 + p], s3 = Sb[1536 + p];
      float mx = fmaxf(fmaxf(s0, s1), fmaxf(s2, s3));
      float e0 = __expf(s0 - mx), e1 = __expf(s1 - mx);
      float e2 = __expf(s2 - mx), e3 = __expf(s3 - mx);
      float inv = 1.0f / (e0 + e1 + e2 + e3);
      float p0 = e0 * inv, p1 = e1 * inv, p2 = e2 * inv, p3 = e3 * inv;
      float mx2 = fmaxf(fmaxf(p0, p1), fmaxf(p2, p3));
      float t0 = __expf(p0 - mx2), t1 = __expf(p1 - mx2);
      float t2 = __expf(p2 - mx2), t3 = __expf(p3 - mx2);
      float inv2 = 1.0f / (t0 + t1 + t2 + t3);
      Wt[p]        = (__bf16)(t0 * inv2);
      Wt[512 + p]  = (__bf16)(t1 * inv2);
      Wt[1024 + p] = (__bf16)(t2 * inv2);
      Wt[1536 + p] = (__bf16)(t3 * inv2);
    }
    __syncthreads();

    // ====== GEMM2: out[b, ntile, :] += W[b] (16x32) @ V[b] (32x128) =======
    const int dw = w * 32;                          // wave owns 32 d-columns
    for (int b = 0; b < BATCH; ++b) {
      // async-stage V tile (32 x 128 f32 = 16 KB) into LDS
      {
        u32 dst = lds_addr32(Vst);
        uint64_t src = (uint64_t)(uintptr_t)(vin + ((size_t)b * SEQ_M + m0) * DIM);
        #pragma unroll
        for (int j = 0; j < 8; ++j) {               // 1024 x 16B / 128 thr
          u32 dlo = dst + (u32)((tid + j * THREADS) * 16);
          uint64_t sa = src + (uint64_t)(tid + j * THREADS) * 16u;
          asm volatile("global_load_async_to_lds_b128 %0, %1, off"
                       :: "v"(dlo), "v"(sa) : "memory");
        }
        asm volatile("s_wait_asynccnt 0x0" ::: "memory");
      }
      __syncthreads();

      v16bf aw = afrag_lds(Wt + b * (NT * MT), MT, r, 0, kbA);
      #pragma unroll
      for (int nc = 0; nc < 2; ++nc) {
        int dcol = dw + nc * 16 + r;
        FragBF fb;                                  // B[K=m][N=d] from LDS cols
        #pragma unroll
        for (int ee = 0; ee < 16; ++ee)
          fb.e[ee] = (__bf16)Vst[(kbB + ee) * DIM + dcol];
        oacc[b][nc].v = __builtin_amdgcn_wmma_f32_16x16x32_bf16(
            false, aw, false, fb.v, (short)0, oacc[b][nc].v, false, false);
      }
      __syncthreads();                              // protect Vst reuse
    }
  }

  // ---- epilogue: C layout -> out[b][n0+row][dw + nc*16 + r] ----
  #pragma unroll
  for (int b = 0; b < BATCH; ++b)
    #pragma unroll
    for (int nc = 0; nc < 2; ++nc)
      #pragma unroll
      for (int rr = 0; rr < 8; ++rr)
        out[((size_t)b * SEQ_N + n0 + rr + 8 * hb) * DIM + w * 32 + nc * 16 + r]
            = oacc[b][nc].e[rr];
}

extern "C" void kernel_launch(void* const* d_in, const int* in_sizes, int n_in,
                              void* d_out, int out_size, void* d_ws, size_t ws_size,
                              hipStream_t stream) {
  (void)in_sizes; (void)n_in; (void)d_ws; (void)ws_size; (void)out_size;
  const float* k = (const float*)d_in[0];
  const float* q = (const float*)d_in[1];
  const float* v = (const float*)d_in[2];
  float* out = (float*)d_out;
  dim3 grid(SEQ_N / NT);   // 256 blocks, one 16-row n-tile each (all batches)
  dim3 block(THREADS);     // 4 wave32s
  hipLaunchKernelGGL(attn_batchsoftmax_kernel, grid, block, 0, stream,
                     k, q, v, out);
}
